// TransformerBlock_65077344469200
// MI455X (gfx1250) — matrix operand order repaired and checked by reading, not compiled, not run on hardware
//
#include <hip/hip_runtime.h>

// ---------------------------------------------------------------------------
// MI455X (gfx1250, wave32) TransformerBlock: all GEMMs via v_wmma_f32_16x16x32_bf16
// (bf16 in, f32 accumulate). Tiles stream into LDS with the CDNA5 async path
// (global_load_async_to_lds_b128 + s_wait_asynccnt), double-buffered so tile
// k+1 copies overlap WMMA on tile k. Elementwise ops fused into GEMM epilogues.
// ---------------------------------------------------------------------------

typedef __bf16 bf16_t;
typedef __attribute__((ext_vector_type(16))) __bf16 v16bf;
typedef __attribute__((ext_vector_type(8)))  float  v8f;

#define N_TOK 8192
#define D_MOD 1024
#define SEQ   2048

union Frag16 { uint4 u[2]; v16bf v; };

enum EpiMode { EPI_BIAS_F32 = 0, EPI_SIG_BF16 = 1, EPI_NBRC = 2 };

__device__ __forceinline__ float dsigmoid(float x) { return 1.f / (1.f + __expf(-x)); }
__device__ __forceinline__ float dgelu(float x) {
  return 0.5f * x * (1.f + erff(x * 0.70710678118654752f));
}

// LDS byte offset of a __shared__ object (generic -> addrspace(3) -> u32).
__device__ __forceinline__ unsigned lds_u32(const void* p) {
  return (unsigned)(unsigned long long)(__attribute__((address_space(3))) const char*)p;
}

// 4x async b128 copies, 64 contiguous bytes global -> LDS. INST_OFFSET is added
// to BOTH the LDS and global address (CDNA5 ISA), so one
// (vdst, vaddr) pair serves all four chunks. Tracked by ASYNCcnt.
#define ASYNC_CP64(ldsoff, gaddr)                                             \
  asm volatile("global_load_async_to_lds_b128 %0, %1, off\n\t"                \
               "global_load_async_to_lds_b128 %0, %1, off offset:16\n\t"      \
               "global_load_async_to_lds_b128 %0, %1, off offset:32\n\t"      \
               "global_load_async_to_lds_b128 %0, %1, off offset:48"          \
               :: "v"(ldsoff),                                                \
                  "v"((unsigned long long)(__UINTPTR_TYPE__)(gaddr))          \
               : "memory")

#define WAIT_ASYNC(n) asm volatile("s_wait_asynccnt " #n ::: "memory")

// --------------------------------------------------------------------------
// C(M x Nout) = A(M x K) * W(Nout x K)^T + bias, bf16 WMMA, f32 accum.
// Block tile 128x128, K-step 64, double-buffered LDS. 8 waves: 2(M) x 4(N),
// wave tile 64x32 (4x2 WMMA frags). All call-site dims are tile multiples.
// --------------------------------------------------------------------------
template<int EPI>
__launch_bounds__(256)
__global__ void gemm_wmma_bf16(const bf16_t* __restrict__ A, int lda,
                               const bf16_t* __restrict__ W, int ldw,
                               const float*  __restrict__ bias,
                               float*        __restrict__ Cf,
                               bf16_t*       __restrict__ Cb,
                               const float*  __restrict__ saP,
                               const float*  __restrict__ scP,
                               const float*  __restrict__ xres,
                               int Nout, int K)
{
  constexpr int BM = 128, BN = 128, BK = 64, LS = BK + 8;  // +8 bf16 pad (bank-safe)
  __shared__ __align__(16) bf16_t As[2][BM * LS];
  __shared__ __align__(16) bf16_t Bs[2][BN * LS];

  const int tid  = threadIdx.x;
  const int lane = tid & 31;
  const int wid  = tid >> 5;
  const int wm   = wid >> 2;          // wave row 0..1  (64 rows each)
  const int wn   = wid & 3;           // wave col 0..3  (32 cols each)
  const int bm   = blockIdx.y * BM;
  const int bn   = blockIdx.x * BN;

  // Cooperative tile copy: 128x64 bf16 = 16KB = 256 thr x 64B (4 async b128).
  const int lrow  = tid >> 1;          // 0..127
  const int lcolb = (tid & 1) << 5;    // 0 / 32 (elements)
  const bf16_t* Ap = A + (size_t)(bm + lrow) * lda + lcolb;
  const bf16_t* Wp = W + (size_t)(bn + lrow) * ldw + lcolb;  // W row-major == B^T

  const unsigned aD0 = lds_u32(&As[0][lrow * LS + lcolb]);
  const unsigned aD1 = lds_u32(&As[1][lrow * LS + lcolb]);
  const unsigned bD0 = lds_u32(&Bs[0][lrow * LS + lcolb]);
  const unsigned bD1 = lds_u32(&Bs[1][lrow * LS + lcolb]);

  v8f acc[4][2];
  {
    v8f z = {0.f, 0.f, 0.f, 0.f, 0.f, 0.f, 0.f, 0.f};
#pragma unroll
    for (int i = 0; i < 4; i++)
#pragma unroll
      for (int j = 0; j < 2; j++) acc[i][j] = z;
  }

  const int half = lane >> 4;   // K-half select per ISA 16-bit A/B VGPR layout
  const int l16  = lane & 15;
  const int nk   = K >> 6;

  // Prologue: tile 0 -> buffer 0.
  ASYNC_CP64(aD0, Ap);
  ASYNC_CP64(bD0, Wp);

  for (int it = 0; it < nk; ++it) {
    if (it + 1 < nk) {                       // stream tile it+1 into other buffer
      const int nxt = (it + 1) & 1;
      const bf16_t* ag = Ap + ((size_t)(it + 1) << 6);
      const bf16_t* bg = Wp + ((size_t)(it + 1) << 6);
      ASYNC_CP64(nxt ? aD1 : aD0, ag);
      ASYNC_CP64(nxt ? bD1 : bD0, bg);
      WAIT_ASYNC(8);                         // in-order: tile `it`'s 8 issues done
    } else {
      WAIT_ASYNC(0);
    }
    __syncthreads();                         // tile `it` visible workgroup-wide

    const bf16_t* Ab = As[it & 1];
    const bf16_t* Bb = Bs[it & 1];
#pragma unroll
    for (int ks = 0; ks < BK; ks += 32) {
      Frag16 a[4], b[2];
      // A frag (16x32): lane<16 -> K{0..7},{16..23}; lane>=16 -> K{8..15},{24..31}
#pragma unroll
      for (int i = 0; i < 4; i++) {
        const bf16_t* p = Ab + (wm * 64 + i * 16 + l16) * LS + ks;
        a[i].u[0] = *(const uint4*)(p + half * 8);
        a[i].u[1] = *(const uint4*)(p + 16 + half * 8);
      }
      // B frag (32x16): lane<16 -> K{0..7},{16..23} of column l16; lane>=16 -> K{8..15},{24..31}
#pragma unroll
      for (int j = 0; j < 2; j++) {
        const bf16_t* p = Bb + (wn * 32 + j * 16 + l16) * LS + ks;
        b[j].u[0] = *(const uint4*)(p + half * 8);
        b[j].u[1] = *(const uint4*)(p + 16 + half * 8);
      }
#pragma unroll
      for (int i = 0; i < 4; i++)
#pragma unroll
        for (int j = 0; j < 2; j++)
          acc[i][j] = __builtin_amdgcn_wmma_f32_16x16x32_bf16(
              false, a[i].v, false, b[j].v, (short)0, acc[i][j], false, false);
    }
    if (it + 1 < nk) __syncthreads();        // all reads done before overwrite
  }

  // Epilogue. C layout: VGPR r -> M = r + 8*half, lane%16 -> N.
#pragma unroll
  for (int i = 0; i < 4; i++) {
#pragma unroll
    for (int j = 0; j < 2; j++) {
      const int col = bn + wn * 32 + j * 16 + l16;
      const float bb = bias[col];
#pragma unroll
      for (int r = 0; r < 8; r++) {
        const int row = bm + wm * 64 + i * 16 + half * 8 + r;
        const size_t idx = (size_t)row * Nout + col;
        const float v = acc[i][j][r] + bb;
        if (EPI == EPI_BIAS_F32) {
          Cf[idx] = v;
        } else if (EPI == EPI_SIG_BF16) {
          Cb[idx] = (bf16_t)dsigmoid(v);
        } else {  // EPI_NBRC: Nout == 1024, fuse full nBRC gate
          const float aa = 1.f + tanhf(saP[idx]);
          const float cc = dsigmoid(scP[idx]);
          const float xv = xres[idx];
          const float o  = cc * xv + (1.f - cc) * tanhf(v + aa * xv);
          Cf[idx] = o;
          Cb[idx] = (bf16_t)o;
        }
      }
    }
  }
}

// ---------------- elementwise / data-prep kernels --------------------------

__global__ void k_rmsnorm(const float* __restrict__ x, const float* __restrict__ g,
                          bf16_t* __restrict__ out)
{
  const int row = blockIdx.x;
  const int tid = threadIdx.x;  // 256 threads, 4 floats each
  const float4 xv = *(const float4*)(x + (size_t)row * D_MOD + tid * 4);
  float s = xv.x * xv.x + xv.y * xv.y + xv.z * xv.z + xv.w * xv.w;
#pragma unroll
  for (int o = 16; o > 0; o >>= 1) s += __shfl_down(s, o, 32);
  __shared__ float red[8];
  if ((tid & 31) == 0) red[tid >> 5] = s;
  __syncthreads();
  if (tid == 0) {
    float t = 0.f;
#pragma unroll
    for (int i = 0; i < 8; i++) t += red[i];
    red[0] = t;
  }
  __syncthreads();
  const float norm = sqrtf(red[0]) * 0.03125f;  // * D^-0.5 (1/32)
  const float inv  = 1.f / fmaxf(norm, 1e-8f);
  const float4 gv = *(const float4*)(g + tid * 4);
  bf16_t* o = out + (size_t)row * D_MOD + tid * 4;
  o[0] = (bf16_t)(xv.x * inv * gv.x);
  o[1] = (bf16_t)(xv.y * inv * gv.y);
  o[2] = (bf16_t)(xv.z * inv * gv.z);
  o[3] = (bf16_t)(xv.w * inv * gv.w);
}

// im2col for k=3, pad=1 conv over each length-2048 sequence: A[row, tap*1024+c]
__global__ void k_im2col(const bf16_t* __restrict__ xn, bf16_t* __restrict__ A)
{
  const int i = blockIdx.x * 256 + threadIdx.x;  // one thread per 8 bf16
  if (i >= N_TOK * 384) return;
  const int row = i / 384;
  const int c8  = (i - row * 384) * 8;
  const int tap = c8 >> 10;
  const int c   = c8 & 1023;
  const int s   = row & (SEQ - 1);
  const int srow = row + tap - 1;
  const bool valid = (tap == 1) | ((tap == 0) & (s > 0)) | ((tap == 2) & (s < SEQ - 1));
  uint4 v = {0u, 0u, 0u, 0u};
  if (valid) v = *(const uint4*)(xn + (size_t)srow * D_MOD + c);
  *(uint4*)(A + (size_t)row * (3 * D_MOD) + c8) = v;
}

// y = v * gelu(gate); pack yx = [bf16(y) | bf16(x)] for the concat-K nBRC GEMMs
__global__ void k_geglu_concat(const float* __restrict__ h, const float* __restrict__ x,
                               bf16_t* __restrict__ yx)
{
  const int i = blockIdx.x * 256 + threadIdx.x;
  if (i >= N_TOK * D_MOD) return;
  const int row = i >> 10, c = i & 1023;
  const size_t hb = (size_t)row * 2048;
  const float v  = h[hb + c];
  const float gt = h[hb + 1024 + c];
  yx[hb + c]        = (bf16_t)(v * dgelu(gt));
  yx[hb + 1024 + c] = (bf16_t)x[i];
}

__global__ void k_cvt(const float* __restrict__ s, bf16_t* __restrict__ d, int n) {
  const int i = blockIdx.x * 256 + threadIdx.x;
  if (i < n) d[i] = (bf16_t)s[i];
}

// conv_w [4d, d, 3] -> bf16 [4d, 3d] with K = tap*1024 + cin (matches im2col)
__global__ void k_cvt_conv(const float* __restrict__ s, bf16_t* __restrict__ d) {
  const int i = blockIdx.x * 256 + threadIdx.x;
  if (i >= 4 * D_MOD * 3 * D_MOD) return;
  const int o   = i / (3 * D_MOD);
  const int rem = i - o * 3 * D_MOD;
  const int tap = rem >> 10;
  const int c   = rem & 1023;
  d[i] = (bf16_t)s[(size_t)o * 3 * D_MOD + c * 3 + tap];
}

// [U | W] -> bf16 [1024, 2048]  (for A-concat nBRC GEMMs)
__global__ void k_concat2(const float* __restrict__ U, const float* __restrict__ W,
                          bf16_t* __restrict__ d) {
  const int i = blockIdx.x * 256 + threadIdx.x;
  if (i >= D_MOD * 2 * D_MOD) return;
  const int r = i >> 11, c = i & 2047;
  d[i] = (bf16_t)((c < D_MOD) ? U[r * D_MOD + c] : W[r * D_MOD + (c - D_MOD)]);
}

__global__ void k_sumw(const float* __restrict__ U, const float* __restrict__ W,
                       bf16_t* __restrict__ d, int n) {
  const int i = blockIdx.x * 256 + threadIdx.x;
  if (i < n) d[i] = (bf16_t)(U[i] + W[i]);
}

__global__ void k_sumb(const float* __restrict__ a, const float* __restrict__ b,
                       float* __restrict__ d, int n) {
  const int i = blockIdx.x * 256 + threadIdx.x;
  if (i < n) d[i] = a[i] + b[i];
}

// ---------------------------------------------------------------------------

extern "C" void kernel_launch(void* const* d_in, const int* in_sizes, int n_in,
                              void* d_out, int out_size, void* d_ws, size_t ws_size,
                              hipStream_t stream)
{
  (void)in_sizes; (void)n_in; (void)out_size; (void)ws_size;
  auto F = [&](int i) { return (const float*)d_in[i]; };
  // dict order: 0=src; ff params at base b: b+0 g, b+1 conv_w, b+2 conv_b,
  // b+3 geglu_w, b+4 geglu_b, then Ua_w,Ua_b,Wa_w,Wa_b,Uc_w,Uc_b,Wc_w,Wc_b,U_w,U_b.
  // bases: ff1=1, attn=16, mlp=26, ff2=36.
  const float* src = F(0);

  char* wsb = (char*)d_ws;
  size_t off = 0;
  auto alloc  = [&](size_t bytes) -> void* {
    off = (off + 255) & ~(size_t)255;
    void* p = wsb + off; off += bytes; return p;
  };
  auto allocB = [&](size_t n) { return (bf16_t*)alloc(n * sizeof(bf16_t)); };
  auto allocF = [&](size_t n) { return (float*)alloc(n * sizeof(float)); };

  struct FFW { bf16_t *convw, *gegluw, *aw, *cw, *uw; float *ab, *cb; };
  struct GW  { bf16_t *aw, *cw, *uw; float *ab, *cb; };
  FFW w1, w2;  GW ga, gm;
  for (FFW* w : {&w1, &w2}) {
    w->convw  = allocB((size_t)4096 * 3072);
    w->gegluw = allocB((size_t)2048 * 4096);
    w->aw = allocB((size_t)1024 * 2048);
    w->cw = allocB((size_t)1024 * 2048);
    w->uw = allocB((size_t)1024 * 1024);
    w->ab = allocF(1024); w->cb = allocF(1024);
  }
  for (GW* w : {&ga, &gm}) {
    w->aw = allocB((size_t)1024 * 1024);
    w->cw = allocB((size_t)1024 * 1024);
    w->uw = allocB((size_t)1024 * 1024);
    w->ab = allocF(1024); w->cb = allocF(1024);
  }

  bf16_t* xn  = allocB((size_t)N_TOK * D_MOD);
  bf16_t* im  = allocB((size_t)N_TOK * 3072);
  bf16_t* a4  = allocB((size_t)N_TOK * 4096);
  float*  h   = allocF((size_t)N_TOK * 2048);
  bf16_t* yx  = allocB((size_t)N_TOK * 2048);
  float*  sa  = allocF((size_t)N_TOK * D_MOD);
  float*  sc  = allocF((size_t)N_TOK * D_MOD);
  float*  x1  = allocF((size_t)N_TOK * D_MOD);
  float*  x2  = allocF((size_t)N_TOK * D_MOD);
  bf16_t* xb1 = allocB((size_t)N_TOK * D_MOD);
  bf16_t* xb2 = allocB((size_t)N_TOK * D_MOD);

  auto grid1 = [](int n) { return dim3((unsigned)((n + 255) / 256)); };
  auto G     = [](int nout) { return dim3((unsigned)(nout / 128), N_TOK / 128); };

  auto cvtFF = [&](int b, FFW& w) {
    k_cvt_conv<<<grid1(4096 * 3072), 256, 0, stream>>>(F(b + 1), w.convw);
    k_cvt<<<grid1(2048 * 4096), 256, 0, stream>>>(F(b + 3), w.gegluw, 2048 * 4096);
    k_concat2<<<grid1(1024 * 2048), 256, 0, stream>>>(F(b + 5), F(b + 7), w.aw);
    k_concat2<<<grid1(1024 * 2048), 256, 0, stream>>>(F(b + 9), F(b + 11), w.cw);
    k_cvt<<<grid1(1024 * 1024), 256, 0, stream>>>(F(b + 13), w.uw, 1024 * 1024);
    k_sumb<<<grid1(1024), 256, 0, stream>>>(F(b + 6), F(b + 8), w.ab, 1024);
    k_sumb<<<grid1(1024), 256, 0, stream>>>(F(b + 10), F(b + 12), w.cb, 1024);
  };
  auto cvtGate = [&](int b, GW& w) {  // nBRC(x,x): x@U^T + x@W^T == x@(U+W)^T
    k_sumw<<<grid1(1024 * 1024), 256, 0, stream>>>(F(b + 0), F(b + 2), w.aw, 1024 * 1024);
    k_sumw<<<grid1(1024 * 1024), 256, 0, stream>>>(F(b + 4), F(b + 6), w.cw, 1024 * 1024);
    k_cvt<<<grid1(1024 * 1024), 256, 0, stream>>>(F(b + 8), w.uw, 1024 * 1024);
    k_sumb<<<grid1(1024), 256, 0, stream>>>(F(b + 1), F(b + 3), w.ab, 1024);
    k_sumb<<<grid1(1024), 256, 0, stream>>>(F(b + 5), F(b + 7), w.cb, 1024);
  };

  auto ff_gate = [&](const float* xin, int b, FFW& w, float* xo, bf16_t* xbo) {
    k_rmsnorm<<<N_TOK, 256, 0, stream>>>(xin, F(b + 0), xn);
    k_im2col<<<grid1(N_TOK * 384), 256, 0, stream>>>(xn, im);
    gemm_wmma_bf16<EPI_SIG_BF16><<<G(4096), 256, 0, stream>>>(
        im, 3072, w.convw, 3072, F(b + 2), nullptr, a4, nullptr, nullptr, nullptr, 4096, 3072);
    gemm_wmma_bf16<EPI_BIAS_F32><<<G(2048), 256, 0, stream>>>(
        a4, 4096, w.gegluw, 4096, F(b + 4), h, nullptr, nullptr, nullptr, nullptr, 2048, 4096);
    k_geglu_concat<<<grid1(N_TOK * D_MOD), 256, 0, stream>>>(h, xin, yx);
    gemm_wmma_bf16<EPI_BIAS_F32><<<G(1024), 256, 0, stream>>>(
        yx, 2048, w.aw, 2048, w.ab, sa, nullptr, nullptr, nullptr, nullptr, 1024, 2048);
    gemm_wmma_bf16<EPI_BIAS_F32><<<G(1024), 256, 0, stream>>>(
        yx, 2048, w.cw, 2048, w.cb, sc, nullptr, nullptr, nullptr, nullptr, 1024, 2048);
    // su = y@U^T + U_b : A = first half of yx rows (lda=2048, K=1024); fused nBRC epilogue
    gemm_wmma_bf16<EPI_NBRC><<<G(1024), 256, 0, stream>>>(
        yx, 2048, w.uw, 1024, F(b + 14), xo, xbo, sa, sc, xin, 1024, 1024);
  };

  auto plain_gate = [&](const float* xin, const bf16_t* xbin, int b, GW& w,
                        float* xo, bf16_t* xbo) {
    gemm_wmma_bf16<EPI_BIAS_F32><<<G(1024), 256, 0, stream>>>(
        xbin, 1024, w.aw, 1024, w.ab, sa, nullptr, nullptr, nullptr, nullptr, 1024, 1024);
    gemm_wmma_bf16<EPI_BIAS_F32><<<G(1024), 256, 0, stream>>>(
        xbin, 1024, w.cw, 1024, w.cb, sc, nullptr, nullptr, nullptr, nullptr, 1024, 1024);
    gemm_wmma_bf16<EPI_NBRC><<<G(1024), 256, 0, stream>>>(
        xbin, 1024, w.uw, 1024, F(b + 9), xo, xbo, sa, sc, xin, 1024, 1024);
  };

  cvtFF(1, w1); cvtGate(16, ga); cvtGate(26, gm); cvtFF(36, w2);

  ff_gate(src, 1, w1, x1, xb1);                       // feed_forward
  plain_gate(x1, xb1, 16, ga, x2, xb2);               // attn gate
  plain_gate(x2, xb2, 26, gm, x1, xb1);               // mlp gate
  ff_gate(x1, 36, w2, (float*)d_out, xb2);            // to_out
}
